// BasicBlock_7541962572126
// MI455X (gfx1250) — compile-verified, matrix operand order checked by reading, not run on hardware
//
#include <hip/hip_runtime.h>

// ---------------------------------------------------------------------------
// Types for CDNA5 WMMA (wave32, 16x16x32 bf16 -> f32)
// ---------------------------------------------------------------------------
typedef __attribute__((ext_vector_type(16))) __bf16        bf16x16;
typedef __attribute__((ext_vector_type(8)))  float         f32x8;
typedef __attribute__((ext_vector_type(4)))  unsigned int  u32x4;
typedef __attribute__((ext_vector_type(2)))  unsigned int  u32x2;
typedef __attribute__((address_space(3)))    unsigned short lds_us;

union Frag { u32x4 q[2]; bf16x16 v; };

__device__ __forceinline__ unsigned short f2bf(float f) {
    unsigned int u = __float_as_uint(f);
    u += 0x7fffu + ((u >> 16) & 1u);          // round-to-nearest-even
    return (unsigned short)(u >> 16);
}
__device__ __forceinline__ float bf2f(unsigned short h) {
    return __uint_as_float(((unsigned int)h) << 16);
}

// Problem constants
#define NB    64
#define CCH   256
#define HH    56
#define WW    56
#define HWP   3136        // 56*56
#define HWP4  784         // HWP/4
#define CHW   802816      // 256*3136
#define NPIX  200704      // 64*3136
#define KDIM  2304        // 256*9
#define NXEL  51380224    // 64*256*3136
#define NWEL  589824      // 256*256*9
#define BNEPS 1e-5f

// ---------------------------------------------------------------------------
// Weight quantization (ternary): absmax -> masked mean -> ternarize to bf16
// ---------------------------------------------------------------------------
__global__ void k_absmax(const float* __restrict__ w, int n, unsigned int* out) {
    unsigned int m = 0;
    for (int i = blockIdx.x * blockDim.x + threadIdx.x; i < n; i += gridDim.x * blockDim.x)
        m = max(m, __float_as_uint(fabsf(w[i])));
    __shared__ unsigned int s[256];
    s[threadIdx.x] = m;
    __syncthreads();
    for (int o = 128; o > 0; o >>= 1) {
        if (threadIdx.x < o) s[threadIdx.x] = max(s[threadIdx.x], s[threadIdx.x + o]);
        __syncthreads();
    }
    if (threadIdx.x == 0) atomicMax(out, s[0]);
}

__global__ void k_wstat(const float* __restrict__ w, int n,
                        const unsigned int* __restrict__ amax,
                        float* sum, unsigned int* cnt) {
    const float th = 0.05f * __uint_as_float(*amax);
    float s = 0.0f;
    unsigned int c = 0;
    for (int i = blockIdx.x * blockDim.x + threadIdx.x; i < n; i += gridDim.x * blockDim.x) {
        float a = fabsf(w[i]);
        if (a >= th) { s += a; c++; }
    }
    __shared__ float sf[256];
    __shared__ unsigned int sc[256];
    sf[threadIdx.x] = s; sc[threadIdx.x] = c;
    __syncthreads();
    for (int o = 128; o > 0; o >>= 1) {
        if (threadIdx.x < o) {
            sf[threadIdx.x] += sf[threadIdx.x + o];
            sc[threadIdx.x] += sc[threadIdx.x + o];
        }
        __syncthreads();
    }
    if (threadIdx.x == 0) { atomicAdd(sum, sf[0]); atomicAdd(cnt, sc[0]); }
}

__global__ void k_wquant(const float* __restrict__ w, int n,
                         const unsigned int* __restrict__ amax,
                         const float* __restrict__ sum,
                         const unsigned int* __restrict__ cnt,
                         unsigned short* __restrict__ wq) {
    const float th = 0.05f * __uint_as_float(*amax);
    unsigned int c = *cnt;
    const float Wv = *sum / (float)max(c, 1u);
    int i = blockIdx.x * blockDim.x + threadIdx.x;
    if (i < n) {
        float v = w[i];
        float q = (v >= th) ? Wv : ((v < -th) ? -Wv : 0.0f);
        wq[i] = f2bf(q);
    }
}

// ---------------------------------------------------------------------------
// f32 -> bf16 conversion of the input tensor (vectorized)
// ---------------------------------------------------------------------------
__global__ void k_f32_to_bf16(const float4* __restrict__ x,
                              unsigned short* __restrict__ o, int n4) {
    for (int i = blockIdx.x * blockDim.x + threadIdx.x; i < n4; i += gridDim.x * blockDim.x) {
        float4 v = x[i];
        u32x2 p;
        p.x = (unsigned int)f2bf(v.x) | ((unsigned int)f2bf(v.y) << 16);
        p.y = (unsigned int)f2bf(v.z) | ((unsigned int)f2bf(v.w) << 16);
        *(u32x2*)(o + (size_t)i * 4) = p;
    }
}

// ---------------------------------------------------------------------------
// Implicit-GEMM 3x3 conv via V_WMMA_F32_16X16X32_BF16
//   GEMM: out[co, pix] = sum_k  Wq[co, k] * im2col(x)[k, pix]
//   Block tile 256(M) x 128(N), K-step 32: the full Cout per block so each
//   pixel's im2col column is gathered exactly once.
//   8 waves as 4(M) x 2(N); wave tile 64x64 -> 4x4 accums, 16 WMMAs/step.
//   Double-buffered LDS; weights staged with global_load_async_to_lds_b128
//   (ASYNCcnt); activations gathered branchlessly (32-bit offsets) to regs,
//   stored to the next buffer after the WMMAs.
// ---------------------------------------------------------------------------
__global__ __launch_bounds__(256) void conv3x3_wmma(
    const unsigned short* __restrict__ xin,   // bf16 NCHW activations
    const unsigned short* __restrict__ wq,    // bf16 [256][2304] weights
    unsigned short* __restrict__ yout)        // bf16 NCHW output
{
    __shared__ __align__(16) unsigned short As[2][256][40];  // [m][k], padded
    __shared__ __align__(16) unsigned short Bs[2][128][40];  // [pix][k], padded

    const int t    = threadIdx.x;
    const int lane = t & 31;
    const int wid  = t >> 5;
    const int wm   = (wid & 3) * 64;     // wave M offset (4 waves in M)
    const int wn   = (wid >> 2) * 64;    // wave N offset (2 waves in N)
    const int lr   = lane & 15;
    const int kh   = (lane >> 4) * 8;    // K half select per ISA layout

    const int p0 = blockIdx.x * 128;

    // ---- B stager: each thread owns one pixel (t>>1) and half the K row ----
    const int bp  = t >> 1;
    const int bks = (t & 1) * 16;
    const int gp  = p0 + bp;
    const int gn  = gp / HWP;
    const int ghw = gp - gn * HWP;
    const int gh  = ghw / WW;
    const int gw  = ghw - gh * WW;
    const unsigned short* xb = xin + ((long)gn * CHW + gh * WW + gw);

    // 3x3 validity mask for this pixel (bit r = tap r in-bounds)
    unsigned vmask = 0;
    #pragma unroll
    for (int r = 0; r < 9; ++r) {
        int ih = gh + r / 3 - 1, iw = gw + r % 3 - 1;
        if ((unsigned)ih < HH && (unsigned)iw < WW) vmask |= 1u << r;
    }
    // incremental (ci, k%9) state for k = step*32 + bks
    int ci0 = (t & 1);        // bks/9  (bks in {0,16})
    int r90 = (t & 1) * 7;    // bks%9

    // ---- A stager: each thread owns one full weight row (64B per K-step) ----
    const int ar = t;
    const unsigned short* awp = wq + (long)ar * KDIM;

    unsigned short bv[16];

    auto issueA = [&](int buf, int k0) {
        // async copy 64B of weights straight into LDS (ASYNCcnt-tracked);
        // INST_OFFSET applies to both LDS and global addresses.
        unsigned aoff = (unsigned)(size_t)(lds_us*)&As[buf][ar][0];
        const unsigned short* gp2 = awp + k0;
        asm volatile("global_load_async_to_lds_b128 %0, %1, off\n\t"
                     "global_load_async_to_lds_b128 %0, %1, off offset:16\n\t"
                     "global_load_async_to_lds_b128 %0, %1, off offset:32\n\t"
                     "global_load_async_to_lds_b128 %0, %1, off offset:48"
                     :: "v"(aoff), "v"(gp2) : "memory");
    };
    auto gatherB = [&]() {
        int cc = ci0, rr = r90;
        #pragma unroll
        for (int j = 0; j < 16; ++j) {
            int rq  = (rr * 11) >> 5;                 // rr/3 for rr<9
            int off = cc * HWP + rr + 53 * rq - 57;   // + (dh-1)*56 + (dw-1)
            // branchless zero-padding: invalid taps read the (always valid)
            // center element and get masked to zero (v_cndmask, no branches)
            unsigned valid = (vmask >> rr) & 1u;
            int a = valid ? off : 0;
            unsigned short v = xb[a];
            bv[j] = valid ? v : (unsigned short)0;
            ++rr; if (rr == 9) { rr = 0; ++cc; }
        }
    };
    auto advance = [&]() { r90 += 5; ci0 += 3; if (r90 >= 9) { r90 -= 9; ++ci0; } };
    auto storeB = [&](int buf) {
        #pragma unroll
        for (int j = 0; j < 16; ++j) Bs[buf][bp][bks + j] = bv[j];
    };

    // ---- prologue: stage tile 0 ----
    issueA(0, 0);
    gatherB(); advance();
    storeB(0);
    asm volatile("s_wait_asynccnt 0x0" ::: "memory");
    __syncthreads();

    f32x8 acc[4][4] = {};
    constexpr int NS = KDIM / 32;

    for (int s = 0; s < NS; ++s) {
        const int cur = s & 1, nxt = cur ^ 1;

        if (s + 1 < NS) {
            issueA(nxt, (s + 1) * 32);
            // branchless prefetch of a future input-channel plane
            int cpf = min(ci0 + 4, CCH - 1);
            __builtin_prefetch(xb + cpf * HWP, 0, 1);
            gatherB(); advance();
        }

        // fragment loads: ISA 16-bit A/B layout -> two contiguous 16B LDS reads
        Frag a[4], b[4];
        #pragma unroll
        for (int mt = 0; mt < 4; ++mt) {
            const unsigned short* pr = &As[cur][wm + mt * 16 + lr][0];
            a[mt].q[0] = *(const u32x4*)(pr + kh);
            a[mt].q[1] = *(const u32x4*)(pr + kh + 16);
        }
        #pragma unroll
        for (int nt = 0; nt < 4; ++nt) {
            const unsigned short* pr = &Bs[cur][wn + nt * 16 + lr][0];
            b[nt].q[0] = *(const u32x4*)(pr + kh);
            b[nt].q[1] = *(const u32x4*)(pr + kh + 16);
        }
        #pragma unroll
        for (int mt = 0; mt < 4; ++mt)
            #pragma unroll
            for (int nt = 0; nt < 4; ++nt)
                acc[mt][nt] = __builtin_amdgcn_wmma_f32_16x16x32_bf16(
                    false, a[mt].v, false, b[nt].v,
                    (short)0, acc[mt][nt], false, false);

        if (s + 1 < NS) storeB(nxt);
        asm volatile("s_wait_asynccnt 0x0" ::: "memory");
        __syncthreads();
    }

    // epilogue: C/D layout -> VGPR g holds M = g + 8*(lane>=16), N = lane%16
    const int mhalf = (lane >> 4) * 8;
    #pragma unroll
    for (int nt = 0; nt < 4; ++nt) {
        int pc = p0 + wn + nt * 16 + lr;
        int n  = pc / HWP;
        int hw = pc - n * HWP;
        long obase = (long)n * CHW + hw;
        #pragma unroll
        for (int mt = 0; mt < 4; ++mt) {
            int co = wm + mt * 16 + mhalf;
            #pragma unroll
            for (int g = 0; g < 8; ++g)
                yout[obase + (long)(co + g) * HWP] = f2bf(acc[mt][nt][g]);
        }
    }
}

// ---------------------------------------------------------------------------
// Per-channel batch statistics over (N, H, W): one block per channel
// ---------------------------------------------------------------------------
__global__ __launch_bounds__(256) void k_chstats(const unsigned short* __restrict__ y,
                                                 float* __restrict__ mean,
                                                 float* __restrict__ var) {
    const int c = blockIdx.x;
    const int t = threadIdx.x;
    float s = 0.0f, s2 = 0.0f;
    for (int n = 0; n < NB; ++n) {
        const u32x2* p = (const u32x2*)(y + (long)n * CHW + (long)c * HWP);
        for (int i = t; i < HWP4; i += 256) {
            u32x2 pk = p[i];
            float v0 = bf2f((unsigned short)(pk.x & 0xffffu));
            float v1 = bf2f((unsigned short)(pk.x >> 16));
            float v2 = bf2f((unsigned short)(pk.y & 0xffffu));
            float v3 = bf2f((unsigned short)(pk.y >> 16));
            s  += (v0 + v1) + (v2 + v3);
            s2 += (v0 * v0 + v1 * v1) + (v2 * v2 + v3 * v3);
        }
    }
    __shared__ float ss[256], sq[256];
    ss[t] = s; sq[t] = s2;
    __syncthreads();
    for (int o = 128; o > 0; o >>= 1) {
        if (t < o) { ss[t] += ss[t + o]; sq[t] += sq[t + o]; }
        __syncthreads();
    }
    if (t == 0) {
        const float inv = 1.0f / (float)(NB * HWP);
        float m = ss[0] * inv;
        mean[c] = m;
        var[c]  = sq[0] * inv - m * m;
    }
}

// ---------------------------------------------------------------------------
// BN + ReLU (bf16 -> bf16), 4 elements/thread
// ---------------------------------------------------------------------------
__global__ void k_bnrelu4(const u32x2* __restrict__ y,
                          const float* __restrict__ mean, const float* __restrict__ var,
                          const float* __restrict__ gamma, const float* __restrict__ beta,
                          u32x2* __restrict__ o, int n4) {
    for (int i = blockIdx.x * blockDim.x + threadIdx.x; i < n4; i += gridDim.x * blockDim.x) {
        int c = (i / HWP4) % CCH;
        float sc = rsqrtf(var[c] + BNEPS) * gamma[c];
        float sb = beta[c] - mean[c] * sc;
        u32x2 pk = y[i];
        unsigned short h0 = (unsigned short)(pk.x & 0xffffu), h1 = (unsigned short)(pk.x >> 16);
        unsigned short h2 = (unsigned short)(pk.y & 0xffffu), h3 = (unsigned short)(pk.y >> 16);
        unsigned int r0 = f2bf(fmaxf(bf2f(h0) * sc + sb, 0.0f));
        unsigned int r1 = f2bf(fmaxf(bf2f(h1) * sc + sb, 0.0f));
        unsigned int r2 = f2bf(fmaxf(bf2f(h2) * sc + sb, 0.0f));
        unsigned int r3 = f2bf(fmaxf(bf2f(h3) * sc + sb, 0.0f));
        u32x2 ov; ov.x = r0 | (r1 << 16); ov.y = r2 | (r3 << 16);
        o[i] = ov;
    }
}

// ---------------------------------------------------------------------------
// Final: BN2 + residual add + ReLU -> f32 output, 4 elements/thread
// ---------------------------------------------------------------------------
__global__ void k_final4(const u32x2* __restrict__ y,
                         const float4* __restrict__ x,
                         const float* __restrict__ mean, const float* __restrict__ var,
                         const float* __restrict__ gamma, const float* __restrict__ beta,
                         float4* __restrict__ out, int n4) {
    for (int i = blockIdx.x * blockDim.x + threadIdx.x; i < n4; i += gridDim.x * blockDim.x) {
        int c = (i / HWP4) % CCH;
        float sc = rsqrtf(var[c] + BNEPS) * gamma[c];
        float sb = beta[c] - mean[c] * sc;
        u32x2 pk = y[i];
        float4 xv = x[i];
        float4 r;
        r.x = fmaxf(bf2f((unsigned short)(pk.x & 0xffffu)) * sc + sb + xv.x, 0.0f);
        r.y = fmaxf(bf2f((unsigned short)(pk.x >> 16))     * sc + sb + xv.y, 0.0f);
        r.z = fmaxf(bf2f((unsigned short)(pk.y & 0xffffu)) * sc + sb + xv.z, 0.0f);
        r.w = fmaxf(bf2f((unsigned short)(pk.y >> 16))     * sc + sb + xv.w, 0.0f);
        out[i] = r;
    }
}

// ---------------------------------------------------------------------------
// Launcher
// ---------------------------------------------------------------------------
extern "C" void kernel_launch(void* const* d_in, const int* in_sizes, int n_in,
                              void* d_out, int out_size, void* d_ws, size_t ws_size,
                              hipStream_t stream) {
    const float* x  = (const float*)d_in[0];
    const float* w1 = (const float*)d_in[1];
    const float* g1 = (const float*)d_in[2];
    const float* b1 = (const float*)d_in[3];
    const float* w2 = (const float*)d_in[4];
    const float* g2 = (const float*)d_in[5];
    const float* b2 = (const float*)d_in[6];

    char* ws = (char*)d_ws;
    // two big bf16 regions (ping/pong), quantized weights, stat scalars
    unsigned short* R0  = (unsigned short*)(ws);                 // x_bf16, then act1
    unsigned short* R1  = (unsigned short*)(ws + 102760448);     // y1, then y2
    unsigned short* wq1 = (unsigned short*)(ws + 205520896);
    unsigned short* wq2 = (unsigned short*)(ws + 206700544);
    char* sc = ws + 207880192;
    unsigned int* amax1 = (unsigned int*)(sc + 0);
    unsigned int* cnt1  = (unsigned int*)(sc + 4);
    float*        sum1  = (float*)       (sc + 8);
    unsigned int* amax2 = (unsigned int*)(sc + 12);
    unsigned int* cnt2  = (unsigned int*)(sc + 16);
    float*        sum2  = (float*)       (sc + 20);
    float* mean1 = (float*)(sc + 1024);
    float* var1  = (float*)(sc + 2048);
    float* mean2 = (float*)(sc + 3072);
    float* var2  = (float*)(sc + 4096);

    // atomically-accumulated scalars must be re-zeroed every launch
    hipMemsetAsync(sc, 0, 24, stream);

    // weight quantization
    k_absmax<<<256, 256, 0, stream>>>(w1, NWEL, amax1);
    k_absmax<<<256, 256, 0, stream>>>(w2, NWEL, amax2);
    k_wstat <<<256, 256, 0, stream>>>(w1, NWEL, amax1, sum1, cnt1);
    k_wstat <<<256, 256, 0, stream>>>(w2, NWEL, amax2, sum2, cnt2);
    k_wquant<<<(NWEL + 255) / 256, 256, 0, stream>>>(w1, NWEL, amax1, sum1, cnt1, wq1);
    k_wquant<<<(NWEL + 255) / 256, 256, 0, stream>>>(w2, NWEL, amax2, sum2, cnt2, wq2);

    // activations -> bf16
    k_f32_to_bf16<<<(NXEL / 4 + 255) / 256, 256, 0, stream>>>((const float4*)x, R0, NXEL / 4);

    const int cgrid = NPIX / 128;   // 1568 blocks, full Cout per block
    const int N4 = NXEL / 4;

    // conv1 -> BN1 stats -> BN1+ReLU
    conv3x3_wmma<<<cgrid, 256, 0, stream>>>(R0, wq1, R1);
    k_chstats<<<CCH, 256, 0, stream>>>(R1, mean1, var1);
    k_bnrelu4<<<(N4 + 255) / 256, 256, 0, stream>>>((const u32x2*)R1, mean1, var1, g1, b1,
                                                    (u32x2*)R0, N4);

    // conv2 -> BN2 stats -> BN2 + residual + ReLU
    conv3x3_wmma<<<cgrid, 256, 0, stream>>>(R0, wq2, R1);
    k_chstats<<<CCH, 256, 0, stream>>>(R1, mean2, var2);
    k_final4<<<(N4 + 255) / 256, 256, 0, stream>>>((const u32x2*)R1, (const float4*)x,
                                                   mean2, var2, g2, b2,
                                                   (float4*)d_out, N4);
}
